// AudioEncoder_14053132992866
// MI455X (gfx1250) — compile-verified, matrix operand order checked by reading, not run on hardware
//
#include <hip/hip_runtime.h>
#include <hip/hip_bf16.h>
#include <math.h>

// ---------------------------------------------------------------------------
// Whisper-style audio encoder for MI455X (gfx1250, wave32, WMMA).
// All GEMMs (convs via im2col, QKV/O, MLP) and both attention matmuls run on
// v_wmma_f32_16x16x32_bf16 with f32 accumulation.  GEMMs stage tiles into LDS
// with CDNA5 async DMA (global_load_async_to_lds_b128 + s_wait_asynccnt),
// double-buffered; wave tile 64x32 = 8 wmma per K-step from 6 fragment loads.
// Flash attention uses v_permlane16_b32 butterflies for softmax reductions.
// ---------------------------------------------------------------------------

typedef __attribute__((ext_vector_type(8)))  float  v8f;
typedef __attribute__((ext_vector_type(8)))  __bf16 v8bf;
typedef __attribute__((ext_vector_type(16))) __bf16 v16bf;

constexpr int BSZ = 2;
constexpr int T0  = 3000;
constexpr int TP1 = 3008;          // padded T for conv1 rows (6016 total = 47*128)
constexpr int NM  = 128;
constexpr int DM  = 1024;
constexpr int D4  = 4096;
constexpr int NH  = 16;
constexpr int DHD = 64;
constexpr int NL  = 4;
constexpr int TC  = 1500;
constexpr int TCP = 1536;          // padded Tc (3072 total = 24*128)
constexpr int MP1 = BSZ * TP1;     // 6016
constexpr int MP2 = BSZ * TCP;     // 3072
constexpr float QK_SCALE = 0.35355339059327373f; // 64^-0.25

// ---------------------------------------------------------------------------
// Device helpers
// ---------------------------------------------------------------------------
__device__ __forceinline__ float gelu_f(float x) {
    return 0.5f * x * (1.0f + erff(x * 0.70710678118654752f));
}

__device__ __forceinline__ v8f zero8() {
    v8f z;
#pragma unroll
    for (int i = 0; i < 8; ++i) z[i] = 0.0f;
    return z;
}

// Load one 16-lane-striped A/B fragment (16x32 bf16) from a row-major matrix.
// p must already point at  base + row*ldk + 8*(lane>>4)  (+k0).
// lo covers K = 8*half .. 8*half+7, hi covers K = 16+8*half .. 16+8*half+7.
__device__ __forceinline__ v16bf load_frag(const __bf16* p) {
    v8bf lo = *(const v8bf*)(p);
    v8bf hi = *(const v8bf*)(p + 16);
    v16bf r;
#pragma unroll
    for (int i = 0; i < 8; ++i) { r[i] = lo[i]; r[i + 8] = hi[i]; }
    return r;
}

__device__ __forceinline__ v8f wmma_bf16(v16bf a, v16bf b, v8f c) {
    return __builtin_amdgcn_wmma_f32_16x16x32_bf16(
        /*neg_a=*/false, a, /*neg_b=*/false, b,
        /*c_mod=*/(short)0, c, /*reuse_a=*/false, /*reuse_b=*/false);
}

// LDS byte offset of a generic pointer to __shared__ data: the flat->LDS
// aperture translation uses addr[31:0] (ISA 10.2), so truncation is exact.
__device__ __forceinline__ uint32_t lds_off_of(const void* p) {
    return (uint32_t)(uintptr_t)p;
}

// CDNA5 async DMA: global (SADDR + 32-bit voffset) -> LDS, 16 bytes.
// Tracked by ASYNCcnt; completes in-order with other async loads of the wave.
__device__ __forceinline__ void async_b128(uint32_t lds, uint32_t voff, const void* base) {
    asm volatile("global_load_async_to_lds_b128 %0, %1, %2"
                 :: "v"(lds), "v"(voff), "s"((uint64_t)(uintptr_t)base)
                 : "memory");
}

// VALU crosslane butterfly within each 16-lane row: lane j reads lane j^ (1<<S).
// v_permlane16_b32 nibble-select tables (nibble j = j ^ mask).
template <int S>
__device__ __forceinline__ float permx(float v) {
    unsigned lo = (S == 0) ? 0x67452301u : (S == 1) ? 0x54761032u
                : (S == 2) ? 0x32107654u : 0xFEDCBA98u;
    unsigned hi = (S == 0) ? 0xEFCDAB89u : (S == 1) ? 0xDCFE98BAu
                : (S == 2) ? 0xBA98FEDCu : 0x76543210u;
    unsigned s = __float_as_uint(v);
    return __uint_as_float(__builtin_amdgcn_permlane16(s, s, lo, hi, false, false));
}

__device__ __forceinline__ float rowmax16(float v) {
    v = fmaxf(v, permx<0>(v));
    v = fmaxf(v, permx<1>(v));
    v = fmaxf(v, permx<2>(v));
    v = fmaxf(v, permx<3>(v));
    return v;
}

__device__ __forceinline__ float rowsum16(float v) {
    v += permx<0>(v);
    v += permx<1>(v);
    v += permx<2>(v);
    v += permx<3>(v);
    return v;
}

// ---------------------------------------------------------------------------
// GEMM with fused epilogues.  C[m][n] = sum_k A[m][k] * Bt[n][k]
// EPI: 0 conv1->gelu->bf16 scatter [b][n][t]
//      1 conv2->gelu->f32 h + pos
//      2 Q  proj (+bias)*scale -> bf16 [b][h][t][dh]
//      3 K  proj        *scale -> bf16 [b][h][t][dh]
//      4 V  proj (+bias)       -> bf16 [b][h][dh][t]   (transposed)
//      5 residual add:  h += acc + bias
//      6 MLP1: gelu -> bf16 [m][4096]
// ---------------------------------------------------------------------------
struct EpiArgs {
    const float* bias;
    const float* pos;
    float*       hres;
    __bf16*      obf;
    float        scale;
};

template <int EPI>
__device__ __forceinline__ void epilogue(int m, int n, float acc, const EpiArgs& ea) {
    if (EPI == 0) {
        int b = m / TP1, t = m - b * TP1;
        if (t < T0)
            ea.obf[((size_t)(b * DM + n)) * T0 + t] = (__bf16)gelu_f(acc + ea.bias[n]);
    } else if (EPI == 1) {
        int b = m / TCP, t = m - b * TCP;
        if (t < TC)
            ea.hres[(size_t)m * DM + n] = gelu_f(acc + ea.bias[n]) + ea.pos[(size_t)t * DM + n];
    } else if (EPI == 2) {
        int b = m / TCP, t = m - b * TCP;
        int hh = n >> 6, dh = n & 63;
        ea.obf[(((size_t)(b * NH + hh)) * TCP + t) * DHD + dh] =
            (__bf16)((acc + ea.bias[n]) * ea.scale);
    } else if (EPI == 3) {
        int b = m / TCP, t = m - b * TCP;
        int hh = n >> 6, dh = n & 63;
        ea.obf[(((size_t)(b * NH + hh)) * TCP + t) * DHD + dh] = (__bf16)(acc * ea.scale);
    } else if (EPI == 4) {
        int b = m / TCP, t = m - b * TCP;
        int hh = n >> 6, dh = n & 63;
        ea.obf[(((size_t)(b * NH + hh)) * DHD + dh) * TCP + t] = (__bf16)(acc + ea.bias[n]);
    } else if (EPI == 5) {
        int t = m % TCP;
        if (t < TC) ea.hres[(size_t)m * DM + n] += acc + ea.bias[n];
    } else if (EPI == 6) {
        ea.obf[(size_t)m * D4 + n] = (__bf16)gelu_f(acc + ea.bias[n]);
    }
}

// Block: 256 threads = 8 waves in a 2(M) x 4(N) arrangement.
// Block tile: 128(M) x 128(N); wave tile: 64x32 (4x2 WMMA fragments):
// 8 wmma per K-step from 6 fragment loads (A-frags reused over both B-frags).
// K-step 32, double-buffered LDS tiles filled by async DMA:
//   per stage each thread issues 4x b128 (2 for A, 2 for B) = 16 KB/block.
template <int EPI>
__global__ __launch_bounds__(256) void gemm_bf16(const __bf16* __restrict__ A,
                                                 const __bf16* __restrict__ Bt,
                                                 int K, EpiArgs ea) {
    __shared__ __align__(16) __bf16 sA[2][128 * 32];   // 8 KB per buffer
    __shared__ __align__(16) __bf16 sB[2][128 * 32];   // 8 KB per buffer

    int tid  = threadIdx.x;
    int lane = tid & 31;
    int w    = tid >> 5;
    int sub  = lane >> 4;   // 16-lane half
    int r16  = lane & 15;
    int M0 = blockIdx.x * 128;
    int N0 = blockIdx.y * 128;
    int wm = w >> 2;        // 0..1 -> 64-row slab
    int wn = w & 3;         // 0..3 -> 32-col slab

    // ---- async loader mapping: rows tid/4 and tid/4+64, 16B chunk tid%4 ----
    int lrow = tid >> 2, lchk = tid & 3;
    uint32_t aoff0 = (uint32_t)(((size_t)(M0 + lrow)      * K + lchk * 8) * 2);
    uint32_t aoff1 = (uint32_t)(((size_t)(M0 + 64 + lrow) * K + lchk * 8) * 2);
    uint32_t boff0 = (uint32_t)(((size_t)(N0 + lrow)      * K + lchk * 8) * 2);
    uint32_t boff1 = (uint32_t)(((size_t)(N0 + 64 + lrow) * K + lchk * 8) * 2);
    uint32_t alds0[2], alds1[2], blds0[2], blds1[2];
#pragma unroll
    for (int s = 0; s < 2; ++s) {
        alds0[s] = lds_off_of(&sA[s][lrow * 32 + lchk * 8]);
        alds1[s] = lds_off_of(&sA[s][(64 + lrow) * 32 + lchk * 8]);
        blds0[s] = lds_off_of(&sB[s][lrow * 32 + lchk * 8]);
        blds1[s] = lds_off_of(&sB[s][(64 + lrow) * 32 + lchk * 8]);
    }

    int nstages = K >> 5;

    // prologue: stage 0 in flight
    async_b128(alds0[0], aoff0, A);
    async_b128(alds1[0], aoff1, A);
    async_b128(blds0[0], boff0, Bt);
    async_b128(blds1[0], boff1, Bt);

    v8f c[4][2];
#pragma unroll
    for (int i = 0; i < 4; ++i) { c[i][0] = zero8(); c[i][1] = zero8(); }

    for (int i = 0; i < nstages; ++i) {
        int cb = i & 1;
        if (i + 1 < nstages) {
            uint32_t kb = (uint32_t)((i + 1) << 6);   // (k0*2) bytes
            int nb = cb ^ 1;
            async_b128(alds0[nb], aoff0 + kb, A);
            async_b128(alds1[nb], aoff1 + kb, A);
            async_b128(blds0[nb], boff0 + kb, Bt);
            async_b128(blds1[nb], boff1 + kb, Bt);
            asm volatile("s_wait_asynccnt %0" :: "i"(4) : "memory");  // stage i arrived
        } else {
            asm volatile("s_wait_asynccnt %0" :: "i"(0) : "memory");
        }
        __syncthreads();   // all waves' DMA data visible

        const __bf16* pa = &sA[cb][(wm * 64 + r16) * 32 + 8 * sub];
        const __bf16* pb = &sB[cb][(wn * 32 + r16) * 32 + 8 * sub];
        v16bf a0 = load_frag(pa);
        v16bf a1 = load_frag(pa + 16 * 32);
        v16bf a2 = load_frag(pa + 32 * 32);
        v16bf a3 = load_frag(pa + 48 * 32);
        v16bf b0 = load_frag(pb);
        v16bf b1 = load_frag(pb + 16 * 32);
        c[0][0] = wmma_bf16(a0, b0, c[0][0]);
        c[0][1] = wmma_bf16(a0, b1, c[0][1]);
        c[1][0] = wmma_bf16(a1, b0, c[1][0]);
        c[1][1] = wmma_bf16(a1, b1, c[1][1]);
        c[2][0] = wmma_bf16(a2, b0, c[2][0]);
        c[2][1] = wmma_bf16(a2, b1, c[2][1]);
        c[3][0] = wmma_bf16(a3, b0, c[3][0]);
        c[3][1] = wmma_bf16(a3, b1, c[3][1]);

        __syncthreads();   // WAR: everyone done reading before buffer is re-filled
    }

#pragma unroll
    for (int i = 0; i < 4; ++i) {
#pragma unroll
        for (int r = 0; r < 8; ++r) {
            int mr = M0 + wm * 64 + 16 * i + r + 8 * sub;
            int nc = N0 + wn * 32 + r16;
            epilogue<EPI>(mr, nc,      c[i][0][r], ea);
            epilogue<EPI>(mr, nc + 16, c[i][1][r], ea);
        }
    }
}

// ---------------------------------------------------------------------------
// Flash attention: one wave per (b,h, 16-query tile).  Streams keys in tiles
// of 32 with online softmax; P goes through LDS to convert C-layout -> A-layout.
// q,k: [b][h][t][dh] bf16 (pre-scaled by DH^-0.25);  vt: [b][h][dh][t] bf16.
// out wv: [b*TCP + t][D] bf16 (A-matrix for the O projection GEMM).
// ---------------------------------------------------------------------------
__global__ __launch_bounds__(32) void attn_kernel(const __bf16* __restrict__ q,
                                                  const __bf16* __restrict__ kmat,
                                                  const __bf16* __restrict__ vt,
                                                  const int* __restrict__ xlen,
                                                  __bf16* __restrict__ wv) {
    __shared__ __align__(16) __bf16 pbuf[16 * 40];   // 16 rows x 32 cols, stride 40

    int lane = threadIdx.x;
    int sub  = lane >> 4;
    int r16  = lane & 15;
    int bh = blockIdx.y;
    int b  = bh >> 4;                 // NH == 16
    int hh = bh & 15;
    int qt = blockIdx.x * 16;
    int xl = xlen[b];

    const __bf16* qbase = q + ((size_t)bh * TCP + qt + r16) * DHD + 8 * sub;
    v16bf qa0 = load_frag(qbase);        // K = 0..31  of DH
    v16bf qa1 = load_frag(qbase + 32);   // K = 32..63 of DH

    v8f o0 = zero8(), o1 = zero8(), o2 = zero8(), o3 = zero8();
    float mrow[8], lrow[8];
#pragma unroll
    for (int r = 0; r < 8; ++r) { mrow[r] = -1e30f; lrow[r] = 0.0f; }

    for (int s0 = 0; s0 < TCP; s0 += 32) {
        v8f sc[2];
        sc[0] = zero8(); sc[1] = zero8();
#pragma unroll
        for (int j2 = 0; j2 < 2; ++j2) {
            const __bf16* kbase = kmat + ((size_t)bh * TCP + s0 + 16 * j2 + r16) * DHD + 8 * sub;
            v16bf kb0 = load_frag(kbase);
            v16bf kb1 = load_frag(kbase + 32);
            sc[j2] = wmma_bf16(qa0, kb0, sc[j2]);
            sc[j2] = wmma_bf16(qa1, kb1, sc[j2]);
            int key = s0 + 16 * j2 + r16;     // this lane's column (key index)
            float kbias = (key < TC && (2 * key + 1) < xl) ? 0.0f : -1e10f;
#pragma unroll
            for (int r = 0; r < 8; ++r) sc[j2][r] += kbias;
        }
        // row-wise max across the 16 lanes of each half (VALU permlane butterflies)
        float nm[8];
#pragma unroll
        for (int r = 0; r < 8; ++r) {
            float v = rowmax16(fmaxf(sc[0][r], sc[1][r]));
            nm[r] = fmaxf(mrow[r], v);
        }
#pragma unroll
        for (int r = 0; r < 8; ++r) {
            float alpha = __expf(mrow[r] - nm[r]);
            mrow[r] = nm[r];
            lrow[r] *= alpha;
            o0[r] *= alpha; o1[r] *= alpha; o2[r] *= alpha; o3[r] *= alpha;
        }
        __syncthreads();   // protect pbuf against previous iteration's readers
#pragma unroll
        for (int r = 0; r < 8; ++r) {
            float p0 = __expf(sc[0][r] - mrow[r]);
            float p1 = __expf(sc[1][r] - mrow[r]);
            lrow[r] += rowsum16(p0 + p1);
            pbuf[(r + 8 * sub) * 40 + r16]      = (__bf16)p0;   // C-layout scatter
            pbuf[(r + 8 * sub) * 40 + 16 + r16] = (__bf16)p1;
        }
        __syncthreads();
        // P as A-fragment (16 queries x 32 keys)
        v16bf pf = load_frag(&pbuf[r16 * 40 + 8 * sub]);
        // PV: B-fragments from transposed V, 4 dh-chunks of 16
        o0 = wmma_bf16(pf, load_frag(vt + ((size_t)bh * DHD +  0 + r16) * TCP + s0 + 8 * sub), o0);
        o1 = wmma_bf16(pf, load_frag(vt + ((size_t)bh * DHD + 16 + r16) * TCP + s0 + 8 * sub), o1);
        o2 = wmma_bf16(pf, load_frag(vt + ((size_t)bh * DHD + 32 + r16) * TCP + s0 + 8 * sub), o2);
        o3 = wmma_bf16(pf, load_frag(vt + ((size_t)bh * DHD + 48 + r16) * TCP + s0 + 8 * sub), o3);
    }
#pragma unroll
    for (int r = 0; r < 8; ++r) {
        float inv = 1.0f / lrow[r];
        int t = qt + r + 8 * sub;
        size_t mg = ((size_t)b * TCP + t) * DM + hh * 64;
        wv[mg +  0 + r16] = (__bf16)(o0[r] * inv);
        wv[mg + 16 + r16] = (__bf16)(o1[r] * inv);
        wv[mg + 32 + r16] = (__bf16)(o2[r] * inv);
        wv[mg + 48 + r16] = (__bf16)(o3[r] * inv);
    }
}

// ---------------------------------------------------------------------------
// LayerNorm: h (f32, padded rows) -> Y (bf16).  Pad rows are zero-filled so
// every padded GEMM/attention row stays finite.
// ---------------------------------------------------------------------------
__global__ __launch_bounds__(256) void ln_kernel(const float* __restrict__ h,
                                                 const float* __restrict__ g,
                                                 const float* __restrict__ be,
                                                 __bf16* __restrict__ Y) {
    int m = blockIdx.x;
    int t = m % TCP;
    __bf16* yrow = Y + (size_t)m * DM;
    if (t >= TC) {
        for (int d = threadIdx.x; d < DM; d += 256) yrow[d] = (__bf16)0.0f;
        return;
    }
    __shared__ float red[256];
    const float* row = h + (size_t)m * DM;
    int tid = threadIdx.x;
    float v[4], s = 0.0f;
#pragma unroll
    for (int j = 0; j < 4; ++j) { v[j] = row[tid + 256 * j]; s += v[j]; }
    red[tid] = s; __syncthreads();
    for (int off = 128; off > 0; off >>= 1) { if (tid < off) red[tid] += red[tid + off]; __syncthreads(); }
    float mean = red[0] * (1.0f / 1024.0f); __syncthreads();
    float vs = 0.0f;
#pragma unroll
    for (int j = 0; j < 4; ++j) { float d = v[j] - mean; vs += d * d; }
    red[tid] = vs; __syncthreads();
    for (int off = 128; off > 0; off >>= 1) { if (tid < off) red[tid] += red[tid + off]; __syncthreads(); }
    float rstd = rsqrtf(red[0] * (1.0f / 1024.0f) + 1e-5f);
#pragma unroll
    for (int j = 0; j < 4; ++j) {
        int d = tid + 256 * j;
        yrow[d] = (__bf16)((v[j] - mean) * rstd * g[d] + be[d]);
    }
}

// Final: pair-average over time then LayerNorm -> f32 output.
__global__ __launch_bounds__(256) void final_ln(const float* __restrict__ h,
                                                const float* __restrict__ g,
                                                const float* __restrict__ be,
                                                float* __restrict__ out) {
    int m = blockIdx.x;               // b*750 + i
    int b = m / 750, i = m % 750;
    const float* r0 = h + ((size_t)b * TCP + 2 * i) * DM;
    const float* r1 = r0 + DM;
    __shared__ float red[256];
    int tid = threadIdx.x;
    float v[4], s = 0.0f;
#pragma unroll
    for (int j = 0; j < 4; ++j) { int d = tid + 256 * j; v[j] = 0.5f * (r0[d] + r1[d]); s += v[j]; }
    red[tid] = s; __syncthreads();
    for (int off = 128; off > 0; off >>= 1) { if (tid < off) red[tid] += red[tid + off]; __syncthreads(); }
    float mean = red[0] * (1.0f / 1024.0f); __syncthreads();
    float vs = 0.0f;
#pragma unroll
    for (int j = 0; j < 4; ++j) { float d = v[j] - mean; vs += d * d; }
    red[tid] = vs; __syncthreads();
    for (int off = 128; off > 0; off >>= 1) { if (tid < off) red[tid] += red[tid + off]; __syncthreads(); }
    float rstd = rsqrtf(red[0] * (1.0f / 1024.0f) + 1e-5f);
#pragma unroll
    for (int j = 0; j < 4; ++j) {
        int d = tid + 256 * j;
        out[(size_t)m * DM + d] = (v[j] - mean) * rstd * g[d] + be[d];
    }
}

__global__ void len_kernel(const int* __restrict__ xlen, float* __restrict__ out) {
    int b = threadIdx.x;
    if (b < BSZ) out[(size_t)BSZ * 750 * DM + b] = (float)(((xlen[b] + 1) / 2) / 2);
}

// ---------------------------------------------------------------------------
// Data preparation kernels
// ---------------------------------------------------------------------------
__global__ void im2col1(const float* __restrict__ x, __bf16* __restrict__ A1) {
    long long idx = (long long)blockIdx.x * 256 + threadIdx.x;
    long long total = (long long)MP1 * 384;
    if (idx >= total) return;
    int m = (int)(idx / 384), kk = (int)(idx % 384);
    int b = m / TP1, t = m - b * TP1;
    int c = kk / 3, dk = kk - c * 3;
    int ti = t + dk - 1;
    float v = 0.0f;
    if (t < T0 && ti >= 0 && ti < T0) v = x[((size_t)b * NM + c) * T0 + ti];
    A1[idx] = (__bf16)v;
}

__global__ void im2col2(const __bf16* __restrict__ h1, __bf16* __restrict__ A2) {
    long long idx = (long long)blockIdx.x * 256 + threadIdx.x;
    long long total = (long long)MP2 * 3072;
    if (idx >= total) return;
    int m = (int)(idx / 3072), kk = (int)(idx % 3072);
    int b = m / TCP, t = m - b * TCP;
    int c = kk / 3, dk = kk - c * 3;
    int ti = 2 * t + dk - 1;
    __bf16 v = (__bf16)0.0f;
    if (t < TC && ti >= 0 && ti < T0) v = h1[((size_t)b * DM + c) * T0 + ti];
    A2[idx] = v;
}

__global__ void cvt_copy(const float* __restrict__ in, __bf16* __restrict__ out, int n) {
    int i = blockIdx.x * 256 + threadIdx.x;
    if (i < n) out[i] = (__bf16)in[i];
}

// in [K][N] f32  ->  out [N][K] bf16
__global__ void cvt_transpose(const float* __restrict__ in, __bf16* __restrict__ out,
                              int K, int N) {
    long long i = (long long)blockIdx.x * 256 + threadIdx.x;
    long long total = (long long)K * N;
    if (i >= total) return;
    int n = (int)(i / K), kk = (int)(i % K);
    out[i] = (__bf16)in[(long long)kk * N + n];
}

// ---------------------------------------------------------------------------
// Host side
// ---------------------------------------------------------------------------
extern "C" void kernel_launch(void* const* d_in, const int* in_sizes, int n_in,
                              void* d_out, int out_size, void* d_ws, size_t ws_size,
                              hipStream_t stream) {
    (void)in_sizes; (void)n_in; (void)out_size; (void)ws_size;

    const float* x       = (const float*)d_in[0];
    const int*   xlen    = (const int*)  d_in[1];
    const float* conv1_w = (const float*)d_in[2];
    const float* conv1_b = (const float*)d_in[3];
    const float* conv2_w = (const float*)d_in[4];
    const float* conv2_b = (const float*)d_in[5];
    const float* pos     = (const float*)d_in[6];
    const float* Wq      = (const float*)d_in[7];
    const float* bq      = (const float*)d_in[8];
    const float* Wk      = (const float*)d_in[9];
    const float* Wv      = (const float*)d_in[10];
    const float* bv      = (const float*)d_in[11];
    const float* Wo      = (const float*)d_in[12];
    const float* bo      = (const float*)d_in[13];
    const float* ln1_g   = (const float*)d_in[14];
    const float* ln1_b   = (const float*)d_in[15];
    const float* W1      = (const float*)d_in[16];
    const float* b1      = (const float*)d_in[17];
    const float* W2      = (const float*)d_in[18];
    const float* b2      = (const float*)d_in[19];
    const float* ln2_g   = (const float*)d_in[20];
    const float* ln2_b   = (const float*)d_in[21];
    const float* after_g = (const float*)d_in[22];
    const float* after_b = (const float*)d_in[23];

    char* wsp = (char*)d_ws;
    auto alloc_bf = [&](size_t n) -> __bf16* {
        __bf16* p = (__bf16*)wsp; wsp += ((n * 2) + 255) & ~(size_t)255; return p;
    };
    auto alloc_f = [&](size_t n) -> float* {
        float* p = (float*)wsp; wsp += ((n * 4) + 255) & ~(size_t)255; return p;
    };

    __bf16* A1   = alloc_bf((size_t)MP1 * 384);
    __bf16* H1   = alloc_bf((size_t)BSZ * DM * T0);
    __bf16* A2   = alloc_bf((size_t)MP2 * 3072);
    float*  HRES = alloc_f ((size_t)MP2 * DM);
    __bf16* Y    = alloc_bf((size_t)MP2 * DM);
    __bf16* Qb   = alloc_bf((size_t)MP2 * DM);
    __bf16* Kb   = alloc_bf((size_t)MP2 * DM);
    __bf16* Vt   = alloc_bf((size_t)MP2 * DM);
    __bf16* WVb  = alloc_bf((size_t)MP2 * DM);
    __bf16* U    = alloc_bf((size_t)MP2 * D4);
    __bf16* CW1  = alloc_bf((size_t)DM * 384);
    __bf16* CW2  = alloc_bf((size_t)DM * 3072);
    __bf16* WQT  = alloc_bf((size_t)NL * DM * DM);
    __bf16* WKT  = alloc_bf((size_t)NL * DM * DM);
    __bf16* WVT  = alloc_bf((size_t)NL * DM * DM);
    __bf16* WOT  = alloc_bf((size_t)NL * DM * DM);
    __bf16* W1T  = alloc_bf((size_t)NL * DM * D4);
    __bf16* W2T  = alloc_bf((size_t)NL * D4 * DM);

    // ---- weight conversion (conv weights already [out][K] with K contiguous) ----
    cvt_copy<<<(DM * 384 + 255) / 256, 256, 0, stream>>>(conv1_w, CW1, DM * 384);
    cvt_copy<<<(DM * 3072 + 255) / 256, 256, 0, stream>>>(conv2_w, CW2, DM * 3072);
    for (int l = 0; l < NL; ++l) {
        size_t o = (size_t)l * DM * DM;
        int gdd = (DM * DM + 255) / 256;
        cvt_transpose<<<gdd, 256, 0, stream>>>(Wq + o, WQT + o, DM, DM);
        cvt_transpose<<<gdd, 256, 0, stream>>>(Wk + o, WKT + o, DM, DM);
        cvt_transpose<<<gdd, 256, 0, stream>>>(Wv + o, WVT + o, DM, DM);
        cvt_transpose<<<gdd, 256, 0, stream>>>(Wo + o, WOT + o, DM, DM);
        size_t o4 = (size_t)l * DM * D4;
        int gd4 = (DM * D4 + 255) / 256;
        cvt_transpose<<<gd4, 256, 0, stream>>>(W1 + o4, W1T + o4, DM, D4); // [D][D4]->[D4][D]
        cvt_transpose<<<gd4, 256, 0, stream>>>(W2 + o4, W2T + o4, D4, DM); // [D4][D]->[D][D4]
    }

    // ---- conv stem ----
    im2col1<<<(int)(((long long)MP1 * 384 + 255) / 256), 256, 0, stream>>>(x, A1);
    {
        EpiArgs ea{}; ea.bias = conv1_b; ea.obf = H1;
        gemm_bf16<0><<<dim3(MP1 / 128, DM / 128), 256, 0, stream>>>(A1, CW1, 384, ea);
    }
    im2col2<<<(int)(((long long)MP2 * 3072 + 255) / 256), 256, 0, stream>>>(H1, A2);
    {
        EpiArgs ea{}; ea.bias = conv2_b; ea.hres = HRES; ea.pos = pos;
        gemm_bf16<1><<<dim3(MP2 / 128, DM / 128), 256, 0, stream>>>(A2, CW2, 3072, ea);
    }

    // ---- transformer layers ----
    for (int l = 0; l < NL; ++l) {
        size_t o = (size_t)l * DM * DM;
        size_t o4 = (size_t)l * DM * D4;

        ln_kernel<<<MP2, 256, 0, stream>>>(HRES, ln1_g + l * DM, ln1_b + l * DM, Y);
        {
            EpiArgs ea{}; ea.bias = bq + l * DM; ea.obf = Qb; ea.scale = QK_SCALE;
            gemm_bf16<2><<<dim3(MP2 / 128, DM / 128), 256, 0, stream>>>(Y, WQT + o, DM, ea);
        }
        {
            EpiArgs ea{}; ea.obf = Kb; ea.scale = QK_SCALE;
            gemm_bf16<3><<<dim3(MP2 / 128, DM / 128), 256, 0, stream>>>(Y, WKT + o, DM, ea);
        }
        {
            EpiArgs ea{}; ea.bias = bv + l * DM; ea.obf = Vt;
            gemm_bf16<4><<<dim3(MP2 / 128, DM / 128), 256, 0, stream>>>(Y, WVT + o, DM, ea);
        }
        attn_kernel<<<dim3(TCP / 16, BSZ * NH), 32, 0, stream>>>(Qb, Kb, Vt, xlen, WVb);
        {
            EpiArgs ea{}; ea.bias = bo + l * DM; ea.hres = HRES;
            gemm_bf16<5><<<dim3(MP2 / 128, DM / 128), 256, 0, stream>>>(WVb, WOT + o, DM, ea);
        }
        ln_kernel<<<MP2, 256, 0, stream>>>(HRES, ln2_g + l * DM, ln2_b + l * DM, Y);
        {
            EpiArgs ea{}; ea.bias = b1 + (size_t)l * D4; ea.obf = U;
            gemm_bf16<6><<<dim3(MP2 / 128, D4 / 128), 256, 0, stream>>>(Y, W1T + o4, DM, ea);
        }
        {
            EpiArgs ea{}; ea.bias = b2 + l * DM; ea.hres = HRES;
            gemm_bf16<5><<<dim3(MP2 / 128, DM / 128), 256, 0, stream>>>(U, W2T + o4, D4, ea);
        }
    }

    // ---- final pooling + LN + lengths ----
    final_ln<<<BSZ * 750, 256, 0, stream>>>(HRES, after_g, after_b, (float*)d_out);
    len_kernel<<<1, BSZ, 0, stream>>>(xlen, (float*)d_out);
}
